// GNN_55233279426937
// MI455X (gfx1250) — matrix operand order repaired and checked by reading, not compiled, not run on hardware
//
#include <hip/hip_runtime.h>
#include <hip/hip_fp16.h>
#include <math.h>

// ---------------------------------------------------------------------------
// GNN pipeline for MI455X (gfx1250, wave32):
//   3x GAT(->128) + 2x GCN(->64,->32) + per-graph softmax agg + 2 linear heads
// Dense transforms: v_wmma_f32_16x16x32_f16 fed from LDS tiles staged with
// global_load_async_to_lds_b128 (ASYNCcnt). Edge message passing uses float
// atomics (bandwidth-bound path).
// ---------------------------------------------------------------------------

typedef _Float16 h16v __attribute__((ext_vector_type(16)));
typedef _Float16 h8v  __attribute__((ext_vector_type(8)));
typedef float    f8v  __attribute__((ext_vector_type(8)));

// ---- float <-> ordered-int encoding for atomic max on floats ---------------
__device__ __forceinline__ int   enc_f(float x) { int i = __float_as_int(x); return i >= 0 ? i : (i ^ 0x7fffffff); }
__device__ __forceinline__ float dec_f(int i)   { return __int_as_float(i >= 0 ? i : (i ^ 0x7fffffff)); }

__device__ __forceinline__ void edge_sd(const int* __restrict__ ei, int E, int e, int& s, int& d) {
  if (e < E) { s = ei[e]; d = ei[E + e]; } else { s = e - E; d = e - E; }  // appended self loops
}

// ---------------------------------------------------------------------------
__global__ void k_fill_u32(unsigned* p, unsigned v, int n) {
  int i = blockIdx.x * blockDim.x + threadIdx.x;
  if (i < n) p[i] = v;
}

// Layer 1: h = x @ W1  (K = 1, trivial outer product)
__global__ void k_l1(const float* __restrict__ x, const float* __restrict__ W1,
                     float* __restrict__ h, int N) {
  int i = blockIdx.x * blockDim.x + threadIdx.x;
  if (i >= N * 128) return;
  int n = i >> 7, c = i & 127;
  h[i] = x[n] * W1[c];
}

// Transpose + convert weights to f16: Wt[f][k] = W[k][f]
__global__ void k_wt16(const float* __restrict__ W, _Float16* __restrict__ Wt, int K, int F) {
  int i = blockIdx.x * blockDim.x + threadIdx.x;
  if (i >= K * F) return;
  int f = i / K, k = i % K;
  Wt[i] = (_Float16)W[(size_t)k * F + f];
}

// WMMA GEMM: C[N,F] = A16[N,K] @ Wt[F,K]^T, one wave per 16x16 tile.
// The 16xK A tile and 16xK B tile are each one contiguous 16*K*2-byte block;
// stage both into LDS with global_load_async_to_lds_b128 (coalesced, no VGPR
// round trip), then feed WMMA operand swizzles from LDS:
//   A lane(hi=lane>>4, m=lane&15): halves 0..7 -> K=k0+hi*16+j,
//                                  halves 8..15 -> K=k0+hi*16+8+j
//   B column n = colBase+(lane&15), halves j -> K = k0 + hi*16 + j
//   D: n = lane&15, element r -> row m = r + 8*hi
template <int K, int F>
__global__ void k_gemm(const _Float16* __restrict__ A, const _Float16* __restrict__ Wt,
                       float* __restrict__ C, int N) {
  __shared__ __align__(64) _Float16 sA[16 * K];
  __shared__ __align__(64) _Float16 sB[16 * K];
  const int lane = threadIdx.x;
  const int hi = lane >> 4, m = lane & 15;
  const int rowBase = blockIdx.x * 16, colBase = blockIdx.y * 16;

  // ---- async stage: global -> LDS (tracked on ASYNCcnt; 1 wave => no barrier)
  {
    const char* gA = (const char*)(A + (size_t)rowBase * K);
    const char* gB = (const char*)(Wt + (size_t)colBase * K);
    unsigned ldsA = (unsigned)(unsigned long long)&sA[0];
    unsigned ldsB = (unsigned)(unsigned long long)&sB[0];
    constexpr int BYTES = 16 * K * 2;          // per tile
#pragma unroll
    for (int it = 0; it < BYTES / 512; ++it) { // 512 = 32 lanes * 16B
      int bo = (it * 32 + lane) * 16;
      unsigned la = ldsA + bo;
      unsigned long long ga = (unsigned long long)(gA + bo);
      asm volatile("global_load_async_to_lds_b128 %0, %1, off"
                   :: "v"(la), "v"(ga) : "memory");
      unsigned lb = ldsB + bo;
      unsigned long long gb = (unsigned long long)(gB + bo);
      asm volatile("global_load_async_to_lds_b128 %0, %1, off"
                   :: "v"(lb), "v"(gb) : "memory");
    }
    asm volatile("s_wait_asynccnt 0" ::: "memory");
  }

  f8v acc = {};
#pragma unroll
  for (int k0 = 0; k0 < K; k0 += 32) {
    h8v alo = *(const h8v*)(sA + m * K + k0 + hi * 16);
    h8v ahi = *(const h8v*)(sA + m * K + k0 + hi * 16 + 8);
    h16v a;
#pragma unroll
    for (int j = 0; j < 8; ++j) { a[j] = alo[j]; a[j + 8] = ahi[j]; }
    h16v b = *(const h16v*)(sB + m * K + k0 + hi * 16);
    acc = __builtin_amdgcn_wmma_f32_16x16x32_f16(false, a, false, b, (short)0, acc, false, false);
  }

  if (rowBase + 16 <= N) {                     // uniform fast path: no per-row EXEC churn
#pragma unroll
    for (int r = 0; r < 8; ++r)
      C[(size_t)(rowBase + r + 8 * hi) * F + colBase + m] = acc[r];
  } else {
#pragma unroll
    for (int r = 0; r < 8; ++r) {
      int orow = rowBase + r + 8 * hi;
      if (orow < N) C[(size_t)orow * F + colBase + m] = acc[r];
    }
  }
}

// Per-node attention dots: sS[n] = h[n]@a_src, sD[n] = h[n]@a_dst (wave per node)
__global__ void k_dots(const float* __restrict__ h, int F,
                       const float* __restrict__ as_, const float* __restrict__ ad_,
                       float* __restrict__ sS, float* __restrict__ sD, int N) {
  int lane = threadIdx.x & 31;
  int n = blockIdx.x * 8 + (threadIdx.x >> 5);
  if (n >= N) return;
  const float* hr = h + (size_t)n * F;
  float vs = 0.f, vd = 0.f;
  for (int c = lane; c < F; c += 32) { float v = hr[c]; vs += v * as_[c]; vd += v * ad_[c]; }
  for (int o = 16; o > 0; o >>= 1) { vs += __shfl_down(vs, o, 32); vd += __shfl_down(vd, o, 32); }
  if (lane == 0) { sS[n] = vs; sD[n] = vd; }
}

__global__ void k_edge_max(const int* __restrict__ ei, int E, int ET,
                           const float* __restrict__ sS, const float* __restrict__ sD,
                           int* __restrict__ mKey) {
  int e = blockIdx.x * blockDim.x + threadIdx.x;
  if (e >= ET) return;
  int s, d; edge_sd(ei, E, e, s, d);
  float z = sS[s] + sD[d];
  float lg = z > 0.f ? z : 0.2f * z;           // leaky_relu(0.2)
  atomicMax(mKey + d, enc_f(lg));
}

__global__ void k_edge_exp(const int* __restrict__ ei, int E, int ET,
                           const float* __restrict__ sS, const float* __restrict__ sD,
                           const int* __restrict__ mKey, float* __restrict__ eE,
                           float* __restrict__ sSum) {
  int e = blockIdx.x * blockDim.x + threadIdx.x;
  if (e >= ET) return;
  int s, d; edge_sd(ei, E, e, s, d);
  float z = sS[s] + sD[d];
  float lg = z > 0.f ? z : 0.2f * z;
  float ex = __expf(lg - dec_f(mKey[d]));
  eE[e] = ex;
  atomicAdd(&sSum[d], ex);
}

// GAT message scatter: out[dst] += alpha * h[src], one wave per edge, F=128
__global__ void k_gat_scatter(const int* __restrict__ ei, int E, int ET,
                              const float* __restrict__ h, const float* __restrict__ eE,
                              const float* __restrict__ sSum, float* __restrict__ out) {
  int e = blockIdx.x * 8 + (threadIdx.x >> 5);
  int lane = threadIdx.x & 31;
  if (e >= ET) return;
  int s, d; edge_sd(ei, E, e, s, d);
  const float* __restrict__ hs = h + (size_t)s * 128;
  float* __restrict__ od = out + (size_t)d * 128;
  __builtin_prefetch(hs + lane, 0, 0);
  float alpha = eE[e] / (sSum[d] + 1e-16f);
#pragma unroll
  for (int q = 0; q < 4; ++q) {
    int c = lane + 32 * q;
    atomicAdd(&od[c], alpha * hs[c]);
  }
}

__global__ void k_deg(const int* __restrict__ ei, int E, int ET, float* __restrict__ deg) {
  int e = blockIdx.x * blockDim.x + threadIdx.x;
  if (e >= ET) return;
  int s, d; edge_sd(ei, E, e, s, d);
  (void)s;
  atomicAdd(&deg[d], 1.0f);
}

__global__ void k_dinv(float* __restrict__ deg, int N) {
  int i = blockIdx.x * blockDim.x + threadIdx.x;
  if (i >= N) return;
  float v = deg[i];
  deg[i] = v > 0.f ? rsqrtf(v) : 0.f;
}

// GCN scatter: out[dst] += dinv[src]*dinv[dst] * h[src], wave per edge
__global__ void k_gcn_scatter(const int* __restrict__ ei, int E, int ET,
                              const float* __restrict__ h, int F,
                              const float* __restrict__ dinv, float* __restrict__ out) {
  int e = blockIdx.x * 8 + (threadIdx.x >> 5);
  int lane = threadIdx.x & 31;
  if (e >= ET) return;
  int s, d; edge_sd(ei, E, e, s, d);
  float nrm = dinv[s] * dinv[d];
  const float* __restrict__ hs = h + (size_t)s * F;
  float* __restrict__ od = out + (size_t)d * F;
  for (int c = lane; c < F; c += 32) atomicAdd(&od[c], nrm * hs[c]);
}

// bias + ELU, optionally also emit f16 copy for the next WMMA GEMM
__global__ void k_bias_elu(float* __restrict__ h, const float* __restrict__ bias,
                           int N, int F, _Float16* __restrict__ h16) {
  int i = blockIdx.x * blockDim.x + threadIdx.x;
  if (i >= N * F) return;
  int c = i % F;
  float v = h[i] + bias[c];
  v = v > 0.f ? v : (__expf(v) - 1.f);
  h[i] = v;
  if (h16) h16[i] = (_Float16)v;
}

// ---- per-graph softmax aggregation over batch (F=32) -----------------------
__global__ void k_agg_max(const float* __restrict__ h, const int* __restrict__ batch,
                          const float* __restrict__ tp, int* __restrict__ gKey, int N) {
  int i = blockIdx.x * blockDim.x + threadIdx.x;
  if (i >= N * 32) return;
  int n = i >> 5, c = i & 31;
  float lg = h[i] * tp[0];
  atomicMax(&gKey[batch[n] * 32 + c], enc_f(lg));
}

__global__ void k_agg_sum(const float* __restrict__ h, const int* __restrict__ batch,
                          const float* __restrict__ tp, const int* __restrict__ gKey,
                          float* __restrict__ gS, int N) {
  int i = blockIdx.x * blockDim.x + threadIdx.x;
  if (i >= N * 32) return;
  int n = i >> 5, c = i & 31;
  int b = batch[n] * 32 + c;
  float ex = __expf(h[i] * tp[0] - dec_f(gKey[b]));
  atomicAdd(&gS[b], ex);
}

__global__ void k_agg_w(const float* __restrict__ h, const int* __restrict__ batch,
                        const float* __restrict__ tp, const int* __restrict__ gKey,
                        const float* __restrict__ gS, float* __restrict__ gOut, int N) {
  int i = blockIdx.x * blockDim.x + threadIdx.x;
  if (i >= N * 32) return;
  int n = i >> 5, c = i & 31;
  int b = batch[n] * 32 + c;
  float ex = __expf(h[i] * tp[0] - dec_f(gKey[b]));
  float w = ex / (gS[b] + 1e-16f);
  atomicAdd(&gOut[b], w * h[i]);
}

// Final heads: elu(g@Wl1+bl1) @ Wl2 + bl2 -> log_softmax.  One thread per graph.
__global__ void k_head(const float* __restrict__ gOut,
                       const float* __restrict__ Wl1, const float* __restrict__ bl1,
                       const float* __restrict__ Wl2, const float* __restrict__ bl2,
                       float* __restrict__ out, int B) {
  int b = blockIdx.x * blockDim.x + threadIdx.x;
  if (b >= B) return;
  const float* g = gOut + (size_t)b * 32;
  float y1[16];
#pragma unroll
  for (int j = 0; j < 16; ++j) {
    float s = bl1[j];
    for (int i = 0; i < 32; ++i) s += g[i] * Wl1[i * 16 + j];
    y1[j] = s > 0.f ? s : (__expf(s) - 1.f);
  }
  float y2[2];
#pragma unroll
  for (int k = 0; k < 2; ++k) {
    float s = bl2[k];
    for (int j = 0; j < 16; ++j) s += y1[j] * Wl2[j * 2 + k];
    y2[k] = s;
  }
  float mx = fmaxf(y2[0], y2[1]);
  float ls = mx + __logf(__expf(y2[0] - mx) + __expf(y2[1] - mx));
  out[b * 2 + 0] = y2[0] - ls;
  out[b * 2 + 1] = y2[1] - ls;
}

// ---------------------------------------------------------------------------
static inline unsigned cdiv(long long a, long long b) { return (unsigned)((a + b - 1) / b); }

extern "C" void kernel_launch(void* const* d_in, const int* in_sizes, int n_in,
                              void* d_out, int out_size, void* d_ws, size_t ws_size,
                              hipStream_t stream) {
  // ---- inputs (setup_inputs order) ----
  const float* x    = (const float*)d_in[0];
  const int*   ei   = (const int*)  d_in[1];
  const int*   batch= (const int*)  d_in[2];
  const float* W1   = (const float*)d_in[3];
  const float* a1s  = (const float*)d_in[4];
  const float* a1d  = (const float*)d_in[5];
  const float* b1   = (const float*)d_in[6];
  const float* W2   = (const float*)d_in[7];
  const float* a2s  = (const float*)d_in[8];
  const float* a2d  = (const float*)d_in[9];
  const float* b2   = (const float*)d_in[10];
  const float* W3   = (const float*)d_in[11];
  const float* a3s  = (const float*)d_in[12];
  const float* a3d  = (const float*)d_in[13];
  const float* b3   = (const float*)d_in[14];
  const float* Wg1  = (const float*)d_in[15];
  const float* bg1  = (const float*)d_in[16];
  const float* Wg2  = (const float*)d_in[17];
  const float* bg2  = (const float*)d_in[18];
  const float* tp   = (const float*)d_in[19];
  const float* Wl1  = (const float*)d_in[20];
  const float* bl1  = (const float*)d_in[21];
  const float* Wl2  = (const float*)d_in[22];
  const float* bl2  = (const float*)d_in[23];

  const int N  = in_sizes[0];
  const int E  = in_sizes[1] / 2;
  const int ET = E + N;            // with appended self loops
  const int B  = out_size / 2;

  // ---- workspace carve-out (256B aligned) ----
  size_t off = 0;
  auto alloc = [&](size_t bytes) -> void* {
    void* p = (char*)d_ws + off;
    off += (bytes + 255) & ~(size_t)255;
    return p;
  };
  float*    bufA = (float*)   alloc((size_t)N * 128 * 4);  // GEMM / transform out
  float*    bufB = (float*)   alloc((size_t)N * 128 * 4);  // scatter accumulation
  _Float16* h16  = (_Float16*)alloc((size_t)N * 128 * 2);  // f16 activations for WMMA
  _Float16* wT   = (_Float16*)alloc(128 * 128 * 2);        // f16 transposed weights
  float*    sS   = (float*)   alloc((size_t)N * 4);
  float*    sD   = (float*)   alloc((size_t)N * 4);
  int*      mKey = (int*)     alloc((size_t)N * 4);
  float*    sSum = (float*)   alloc((size_t)N * 4);
  float*    eE   = (float*)   alloc((size_t)ET * 4);
  float*    dinv = (float*)   alloc((size_t)N * 4);
  int*      gKey = (int*)     alloc((size_t)B * 32 * 4);
  float*    gS   = (float*)   alloc((size_t)B * 32 * 4);
  float*    gOut = (float*)   alloc((size_t)B * 32 * 4);
  (void)ws_size; (void)n_in;

  const unsigned T = 256;
  const unsigned gEdges = cdiv(ET, T);
  const unsigned gEdgeW = cdiv(ET, 8);     // wave-per-edge kernels

  // ---- GAT edge pipeline (shared by the 3 GAT layers, F = 128) ----
  auto gat_edges = [&](const float* hin, const float* as_, const float* ad_,
                       const float* bias, float* hout, _Float16* h16out) {
    k_dots<<<cdiv(N, 8), T, 0, stream>>>(hin, 128, as_, ad_, sS, sD, N);
    k_fill_u32<<<cdiv(N, T), T, 0, stream>>>((unsigned*)mKey, 0x80000000u, N);
    k_fill_u32<<<cdiv(N, T), T, 0, stream>>>((unsigned*)sSum, 0u, N);
    k_fill_u32<<<cdiv((size_t)N * 128, T), T, 0, stream>>>((unsigned*)hout, 0u, N * 128);
    k_edge_max<<<gEdges, T, 0, stream>>>(ei, E, ET, sS, sD, mKey);
    k_edge_exp<<<gEdges, T, 0, stream>>>(ei, E, ET, sS, sD, mKey, eE, sSum);
    k_gat_scatter<<<gEdgeW, T, 0, stream>>>(ei, E, ET, hin, eE, sSum, hout);
    k_bias_elu<<<cdiv((size_t)N * 128, T), T, 0, stream>>>(hout, bias, N, 128, h16out);
  };

  // ---- Layer 1: GAT 1 -> 128 (K=1 transform: outer product) ----
  k_l1<<<cdiv((size_t)N * 128, T), T, 0, stream>>>(x, W1, bufA, N);
  gat_edges(bufA, a1s, a1d, b1, bufB, h16);

  // ---- Layer 2: GAT 128 -> 128 (WMMA GEMM) ----
  k_wt16<<<cdiv(128 * 128, T), T, 0, stream>>>(W2, wT, 128, 128);
  k_gemm<128, 128><<<dim3(cdiv(N, 16), 8), 32, 0, stream>>>(h16, wT, bufA, N);
  gat_edges(bufA, a2s, a2d, b2, bufB, h16);

  // ---- Layer 3: GAT 128 -> 128 ----
  k_wt16<<<cdiv(128 * 128, T), T, 0, stream>>>(W3, wT, 128, 128);
  k_gemm<128, 128><<<dim3(cdiv(N, 16), 8), 32, 0, stream>>>(h16, wT, bufA, N);
  gat_edges(bufA, a3s, a3d, b3, bufB, h16);

  // ---- GCN degrees (same graph for both GCN layers) ----
  k_fill_u32<<<cdiv(N, T), T, 0, stream>>>((unsigned*)dinv, 0u, N);
  k_deg<<<gEdges, T, 0, stream>>>(ei, E, ET, dinv);
  k_dinv<<<cdiv(N, T), T, 0, stream>>>(dinv, N);

  // ---- GCN1: 128 -> 64 ----
  k_wt16<<<cdiv(128 * 64, T), T, 0, stream>>>(Wg1, wT, 128, 64);
  k_gemm<128, 64><<<dim3(cdiv(N, 16), 4), 32, 0, stream>>>(h16, wT, bufA, N);
  k_fill_u32<<<cdiv((size_t)N * 64, T), T, 0, stream>>>((unsigned*)bufB, 0u, N * 64);
  k_gcn_scatter<<<gEdgeW, T, 0, stream>>>(ei, E, ET, bufA, 64, dinv, bufB);
  k_bias_elu<<<cdiv((size_t)N * 64, T), T, 0, stream>>>(bufB, bg1, N, 64, h16);  // h16 now [N,64]

  // ---- GCN2: 64 -> 32 ----
  k_wt16<<<cdiv(64 * 32, T), T, 0, stream>>>(Wg2, wT, 64, 32);
  k_gemm<64, 32><<<dim3(cdiv(N, 16), 2), 32, 0, stream>>>(h16, wT, bufA, N);
  k_fill_u32<<<cdiv((size_t)N * 32, T), T, 0, stream>>>((unsigned*)bufB, 0u, N * 32);
  k_gcn_scatter<<<gEdgeW, T, 0, stream>>>(ei, E, ET, bufA, 32, dinv, bufB);
  k_bias_elu<<<cdiv((size_t)N * 32, T), T, 0, stream>>>(bufB, bg2, N, 32, (_Float16*)nullptr);

  // ---- per-graph softmax aggregation (bufB: [N,32]) ----
  k_fill_u32<<<cdiv(B * 32, T), T, 0, stream>>>((unsigned*)gKey, 0x80000000u, B * 32);
  k_fill_u32<<<cdiv(B * 32, T), T, 0, stream>>>((unsigned*)gS, 0u, B * 32);
  k_fill_u32<<<cdiv(B * 32, T), T, 0, stream>>>((unsigned*)gOut, 0u, B * 32);
  k_agg_max<<<cdiv((size_t)N * 32, T), T, 0, stream>>>(bufB, batch, tp, gKey, N);
  k_agg_sum<<<cdiv((size_t)N * 32, T), T, 0, stream>>>(bufB, batch, tp, gKey, gS, N);
  k_agg_w  <<<cdiv((size_t)N * 32, T), T, 0, stream>>>(bufB, batch, tp, gKey, gS, gOut, N);

  // ---- heads + log_softmax ----
  k_head<<<cdiv(B, 64), 64, 0, stream>>>(gOut, Wl1, bl1, Wl2, bl2, (float*)d_out, B);
}
